// SALayer_16561393893689
// MI455X (gfx1250) — matrix operand order repaired and checked by reading, not compiled, not run on hardware
//
#include <hip/hip_runtime.h>
#include <hip/hip_bf16.h>
#include <stdint.h>

// ---------------------------------------------------------------------------
// PointNet++ MSG set-abstraction layer for gfx1250 (MI455X).
// GEMMs on v_wmma_f32_16x16x32_bf16 (fp32 accumulate), register-blocked
// 64x32 per wave (4 m-tiles x 2 n-tiles) for 4x B-operand / 2x A-operand
// reuse. BN+ReLU fused in epilogue; ns-maxpool fused into the last MLP GEMM
// via cross-lane max + integer atomicMax (post-ReLU values >= 0).
// ---------------------------------------------------------------------------

#define NPTS   16384
#define BATCH  4
#define SPTS   1024
#define CIN    128
#define KPAD0  160      // layer-0 K (=131 channels) padded to a multiple of 32
#define EPSV   1e-5f

typedef __attribute__((ext_vector_type(16))) __bf16 bf16x16;
typedef __attribute__((ext_vector_type(8)))  float  f32x8;

__device__ __forceinline__ unsigned short f2bf(float f) {
    unsigned int u = __float_as_uint(f);
    unsigned int r = u + 0x7FFFu + ((u >> 16) & 1u);   // round-to-nearest-even
    return (unsigned short)(r >> 16);
}

// ---------------------------------------------------------------------------
// Furthest point sampling: one block per batch; 1023 dependent argmax steps.
// ---------------------------------------------------------------------------
__global__ void fps_kernel(const float* __restrict__ xyz, float* __restrict__ dists,
                           float* __restrict__ out_newxyz, float* __restrict__ out_idx) {
    const int b   = blockIdx.x;
    const int tid = threadIdx.x;
    const float* p = xyz + (size_t)b * NPTS * 3;
    float* dd = dists + (size_t)b * NPTS;
    __shared__ float sv[256];
    __shared__ int   si[256];
    __shared__ int   s_cur;

    for (int j = tid; j < NPTS; j += 256) dd[j] = 1e10f;
    if (tid == 0) {
        out_newxyz[(size_t)(b * SPTS) * 3 + 0] = p[0];
        out_newxyz[(size_t)(b * SPTS) * 3 + 1] = p[1];
        out_newxyz[(size_t)(b * SPTS) * 3 + 2] = p[2];
        out_idx[b * SPTS] = 0.0f;
        s_cur = 0;
    }
    __syncthreads();
    int cur = 0;
    for (int i = 1; i < SPTS; i++) {
        const float lx = p[cur * 3 + 0], ly = p[cur * 3 + 1], lz = p[cur * 3 + 2];
        float best = -1.0f; int bi = 0;
        for (int j = tid; j < NPTS; j += 256) {
            float dx = p[j * 3 + 0] - lx, dy = p[j * 3 + 1] - ly, dz = p[j * 3 + 2] - lz;
            float d  = dx * dx + dy * dy + dz * dz;
            float nd = fminf(dd[j], d);
            dd[j] = nd;
            if (nd > best) { best = nd; bi = j; }
        }
        sv[tid] = best; si[tid] = bi;
        __syncthreads();
        for (int off = 128; off > 0; off >>= 1) {
            if (tid < off) {
                float ov = sv[tid + off]; int oi = si[tid + off];
                if (ov > sv[tid] || (ov == sv[tid] && oi < si[tid])) { sv[tid] = ov; si[tid] = oi; }
            }
            __syncthreads();
        }
        if (tid == 0) {
            int c = si[0];
            s_cur = c;
            out_newxyz[((size_t)b * SPTS + i) * 3 + 0] = p[c * 3 + 0];
            out_newxyz[((size_t)b * SPTS + i) * 3 + 1] = p[c * 3 + 1];
            out_newxyz[((size_t)b * SPTS + i) * 3 + 2] = p[c * 3 + 2];
            out_idx[b * SPTS + i] = (float)c;
        }
        __syncthreads();
        cur = s_cur;
    }
}

// ---------------------------------------------------------------------------
// Ball query: first nsample in-radius neighbors in index order; pad with first.
// ---------------------------------------------------------------------------
__global__ void ball_query_kernel(const float* __restrict__ xyz, const float* __restrict__ new_xyz,
                                  int* __restrict__ idx_out, float r2, int nsample) {
    int t = blockIdx.x * blockDim.x + threadIdx.x;
    if (t >= BATCH * SPTS) return;
    int b = t / SPTS;
    const float* q = new_xyz + (size_t)t * 3;
    const float qx = q[0], qy = q[1], qz = q[2];
    const float* p = xyz + (size_t)b * NPTS * 3;
    int* out = idx_out + (size_t)t * nsample;
    int cnt = 0, first = 0; bool found = false;
    for (int j = 0; j < NPTS && cnt < nsample; j++) {
        float dx = p[j * 3 + 0] - qx, dy = p[j * 3 + 1] - qy, dz = p[j * 3 + 2] - qz;
        float d = dx * dx + dy * dy + dz * dz;
        if (d < r2) {
            if (!found) { found = true; first = j; }
            out[cnt++] = j;
        }
    }
    for (int k = cnt; k < nsample; k++) out[k] = first;
}

// ---------------------------------------------------------------------------
// fp32 weights (O x C) -> padded bf16 (O x Kpad), zero-fill pad.
// ---------------------------------------------------------------------------
__global__ void wconv_kernel(const float* __restrict__ w, unsigned short* __restrict__ dst,
                             int O, int C, int Kpad) {
    int i = blockIdx.x * blockDim.x + threadIdx.x;
    if (i >= O * Kpad) return;
    int o = i / Kpad, k = i - o * Kpad;
    dst[i] = f2bf((k < C) ? w[o * C + k] : 0.0f);
}

// BN folding: scb[m] = (gamma/sqrt(var+eps), beta - mean*scale)
__global__ void bnprep_kernel(const float* __restrict__ g, const float* __restrict__ b,
                              const float* __restrict__ m, const float* __restrict__ v,
                              float2* __restrict__ scb, int O) {
    int i = blockIdx.x * blockDim.x + threadIdx.x;
    if (i >= O) return;
    float sc = g[i] * rsqrtf(v[i] + EPSV);
    scb[i] = make_float2(sc, b[i] - m[i] * sc);
}

// features (B,C,N) fp32 -> point-major bf16 featT[(b*N+n)*128 + c]
__global__ void ftrans_kernel(const float* __restrict__ feat, unsigned short* __restrict__ featT) {
    int i = blockIdx.x * blockDim.x + threadIdx.x;
    if (i >= BATCH * NPTS * CIN) return;
    int c = i & (CIN - 1);
    int n = (i >> 7) & (NPTS - 1);
    int b = i >> 21;
    featT[i] = f2bf(feat[((size_t)b * CIN + c) * NPTS + n]);
}

// ---------------------------------------------------------------------------
// Gather grouped features in point-major bf16 layout: grouped[col][KPAD0].
// col = s*ns + j ; channels: 0..2 centered xyz, 3..130 features, 131..159 zero.
// ---------------------------------------------------------------------------
__global__ void gather_kernel(const float* __restrict__ xyz, const unsigned short* __restrict__ featT,
                              const float* __restrict__ new_xyz, const int* __restrict__ idx,
                              unsigned short* __restrict__ grouped, int b, int nsample) {
    int col = blockIdx.x;
    int c   = threadIdx.x;
    int s   = col / nsample;
    int j   = col - s * nsample;
    int ip  = idx[((size_t)(b * SPTS + s)) * nsample + j];
    unsigned short v;
    if (c < 3)
        v = f2bf(xyz[((size_t)b * NPTS + ip) * 3 + c] - new_xyz[((size_t)b * SPTS + s) * 3 + c]);
    else if (c < 131)
        v = featT[((size_t)(b * NPTS) + ip) * CIN + (c - 3)];   // contiguous 256B per col
    else
        v = 0;
    grouped[(size_t)col * KPAD0 + c] = v;
}

__global__ void zero_kernel(int* p, int n) {
    int i = blockIdx.x * blockDim.x + threadIdx.x;
    if (i < n) p[i] = 0;
}

__global__ void pool2bf_kernel(const int* __restrict__ pooled, unsigned short* __restrict__ dst, int n) {
    int i = blockIdx.x * blockDim.x + threadIdx.x;
    if (i < n) dst[i] = f2bf(__int_as_float(pooled[i]));
}

// ---------------------------------------------------------------------------
// Register-blocked WMMA GEMM: each wave computes a 64x32 D macro-tile
// (MBLK=4 m-tiles x NBLK=2 n-tiles). B fragments loaded once per K-step and
// reused by 4 WMMAs; A fragments reused by 2. K loop fully unrolled (KPAD is
// a template constant). Fragment maps per CDNA5 ISA 7.12.2:
//   A: lane l<16 -> row mo+l, K e{0..7,16..23}+kt ; lane>=16 -> +8 K offset
//   B: lane l    -> col no+(l&15), K e{0..7,16..23}+kt ; lane>=16 -> +8 K offset
//   C/D: elem r  -> m = mo + r + 8*(l>>4), n = no + (l&15)
// MODE 0: write bf16 activations [col][O] (next layer's B operand)
// MODE 1: fused ns-maxpool -> atomicMax into pooled[(b*S+s)*512 + och_base+m]
// MODE 2: final layer, write fp32 nf (B,512,S) into d_out
// ---------------------------------------------------------------------------
#define MBLK 4
#define NBLK 2

template <int MODE, int KPAD>
__global__ void gemm_kernel(const unsigned short* __restrict__ Abf,
                            const unsigned short* __restrict__ Bact,
                            unsigned short* __restrict__ Cout,
                            int* __restrict__ pooled,
                            float* __restrict__ Fout,
                            const float2* __restrict__ scb,
                            int O, int ncols, int nsample, int och_base, int b) {
    const int lane  = threadIdx.x & 31;
    const int wid   = threadIdx.x >> 5;
    const int no    = (blockIdx.x * (blockDim.x >> 5) + wid) * (16 * NBLK);
    const int mo    = blockIdx.y * (16 * MBLK);
    if (no >= ncols) return;                     // wave-uniform: EXEC stays all-ones
    const int half = lane >> 4;
    const int l15  = lane & 15;

    const unsigned short* arow = Abf  + (size_t)(mo + l15) * KPAD + half * 8;
    const unsigned short* bcol = Bact + (size_t)(no + l15) * KPAD + half * 8;

    f32x8 acc[MBLK][NBLK] = {};
    union F { uint4 q[2]; bf16x16 v; };

#pragma unroll
    for (int kt = 0; kt < KPAD; kt += 32) {
        F bf[NBLK];
#pragma unroll
        for (int ni = 0; ni < NBLK; ni++) {
            const unsigned short* bp = bcol + (size_t)ni * 16 * KPAD + kt;
            bf[ni].q[0] = *(const uint4*)(bp);        // K: kt+half*8    .. +7
            bf[ni].q[1] = *(const uint4*)(bp + 16);   // K: kt+half*8+16 .. +7
        }
#pragma unroll
        for (int mi = 0; mi < MBLK; mi++) {
            F a;
            const unsigned short* ap = arow + (size_t)mi * 16 * KPAD + kt;
            a.q[0] = *(const uint4*)(ap);             // K: kt+half*8    .. +7
            a.q[1] = *(const uint4*)(ap + 16);        // K: kt+half*8+16 .. +7
#pragma unroll
            for (int ni = 0; ni < NBLK; ni++)
                acc[mi][ni] = __builtin_amdgcn_wmma_f32_16x16x32_bf16(
                    false, a.v, false, bf[ni].v, (short)0, acc[mi][ni], false, false);
        }
    }

#pragma unroll
    for (int mi = 0; mi < MBLK; mi++) {
#pragma unroll
        for (int ni = 0; ni < NBLK; ni++) {
            const int n = no + ni * 16 + l15;
            if (MODE == 0) {
                union { unsigned short s[8]; uint4 q; } ov;
#pragma unroll
                for (int r = 0; r < 8; r++) {
                    int m = mo + mi * 16 + half * 8 + r;
                    float2 sb = scb[m];
                    ov.s[r] = f2bf(fmaxf(acc[mi][ni][r] * sb.x + sb.y, 0.0f));
                }
                *(uint4*)(Cout + (size_t)n * O + mo + mi * 16 + half * 8) = ov.q;
            } else if (MODE == 1) {
                int s = (no + ni * 16) / nsample;     // 16-col tile lies in one s-group
#pragma unroll
                for (int r = 0; r < 8; r++) {
                    int m = mo + mi * 16 + half * 8 + r;
                    float2 sb = scb[m];
                    float v = fmaxf(acc[mi][ni][r] * sb.x + sb.y, 0.0f);
                    v = fmaxf(v, __shfl_xor(v, 1, 32));   // reduce 16 lanes sharing m
                    v = fmaxf(v, __shfl_xor(v, 2, 32));
                    v = fmaxf(v, __shfl_xor(v, 4, 32));
                    v = fmaxf(v, __shfl_xor(v, 8, 32));
                    if (l15 == 0)
                        atomicMax(&pooled[((size_t)(b * SPTS + s)) * 512 + och_base + m],
                                  __float_as_int(v));      // exact for v >= 0
                }
            } else {
                int b2 = n >> 10;                     // n = b*S + s
                int s  = n & 1023;
#pragma unroll
                for (int r = 0; r < 8; r++) {
                    int m = mo + mi * 16 + half * 8 + r;
                    float2 sb = scb[m];
                    Fout[((size_t)(b2 * 512 + m)) * SPTS + s] =
                        fmaxf(acc[mi][ni][r] * sb.x + sb.y, 0.0f);
                }
            }
        }
    }
}

// ---------------------------------------------------------------------------
extern "C" void kernel_launch(void* const* d_in, const int* in_sizes, int n_in,
                              void* d_out, int out_size, void* d_ws, size_t ws_size,
                              hipStream_t stream) {
    (void)in_sizes; (void)n_in; (void)out_size; (void)ws_size;
    const float* xyz  = (const float*)d_in[0];
    const float* feat = (const float*)d_in[1];

    float* out        = (float*)d_out;
    float* out_newxyz = out;                                   // (B,S,3)
    float* out_nf     = out + (size_t)BATCH * SPTS * 3;        // (B,512,S)
    float* out_idx    = out_nf + (size_t)BATCH * 512 * SPTS;   // (B,S) idx as float

    // ---- workspace layout (~86 MB) ----
    char* ws = (char*)d_ws;
    size_t off = 0;
    auto alloc = [&](size_t bytes) -> char* {
        char* p = ws + off; off += (bytes + 255) & ~(size_t)255; return p;
    };
    float*          dists    = (float*)alloc((size_t)BATCH * NPTS * sizeof(float));
    int*            idxbuf   = (int*)alloc((size_t)BATCH * SPTS * 64 * sizeof(int));
    const int Osz[7] = {128, 128, 256, 128, 128, 256, 512};
    const int Csz[7] = {131, 128, 128, 131, 128, 128, 512};
    const int Kp [7] = {160, 128, 128, 160, 128, 128, 512};
    unsigned short* wbf[7];
    float2*         scb[7];
    for (int l = 0; l < 7; l++) {
        wbf[l] = (unsigned short*)alloc((size_t)Osz[l] * Kp[l] * 2);
        scb[l] = (float2*)alloc((size_t)Osz[l] * sizeof(float2));
    }
    unsigned short* featT    = (unsigned short*)alloc((size_t)BATCH * NPTS * CIN * 2);
    unsigned short* grouped  = (unsigned short*)alloc((size_t)SPTS * 64 * KPAD0 * 2);
    unsigned short* actA     = (unsigned short*)alloc((size_t)SPTS * 64 * 128 * 2);
    unsigned short* actB     = (unsigned short*)alloc((size_t)SPTS * 64 * 128 * 2);
    int*            pooled   = (int*)alloc((size_t)BATCH * SPTS * 512 * sizeof(int));
    unsigned short* pooledbf = (unsigned short*)alloc((size_t)BATCH * SPTS * 512 * 2);

    // ---- param pointers (pytree order: mlps[2][3](w,g,b,m,v), out(w,g,b,m,v)) ----
    const float *wptr[7], *gm[7], *bt[7], *mn[7], *vr[7];
    for (int si = 0; si < 2; si++)
        for (int li = 0; li < 3; li++) {
            int base = 2 + si * 15 + li * 5;
            int l = si * 3 + li;
            wptr[l] = (const float*)d_in[base + 0];
            gm[l]   = (const float*)d_in[base + 1];
            bt[l]   = (const float*)d_in[base + 2];
            mn[l]   = (const float*)d_in[base + 3];
            vr[l]   = (const float*)d_in[base + 4];
        }
    wptr[6] = (const float*)d_in[32]; gm[6] = (const float*)d_in[33];
    bt[6]   = (const float*)d_in[34]; mn[6] = (const float*)d_in[35];
    vr[6]   = (const float*)d_in[36];

    // ---- 1) FPS (writes new_xyz + fps_idx into d_out) ----
    fps_kernel<<<BATCH, 256, 0, stream>>>(xyz, dists, out_newxyz, out_idx);

    // ---- 2) weight conversion + BN folding + feature transpose ----
    for (int l = 0; l < 7; l++) {
        int ne = Osz[l] * Kp[l];
        wconv_kernel<<<(ne + 255) / 256, 256, 0, stream>>>(wptr[l], wbf[l], Osz[l], Csz[l], Kp[l]);
        bnprep_kernel<<<(Osz[l] + 63) / 64, 64, 0, stream>>>(gm[l], bt[l], mn[l], vr[l], scb[l], Osz[l]);
    }
    ftrans_kernel<<<(BATCH * NPTS * CIN + 255) / 256, 256, 0, stream>>>(feat, featT);

    // ---- 3) pooled buffer init (atomicMax target; 0.0f == int 0) ----
    zero_kernel<<<(BATCH * SPTS * 512 + 255) / 256, 256, 0, stream>>>(pooled, BATCH * SPTS * 512);

    // ---- 4) per-scale, per-batch MLP pipeline ----
    const float radii2[2] = {0.25f, 1.0f};
    const int   nsamp[2]  = {32, 64};
    for (int si = 0; si < 2; si++) {
        const int ns = nsamp[si];
        const int ncols = SPTS * ns;
        ball_query_kernel<<<(BATCH * SPTS + 127) / 128, 128, 0, stream>>>(
            xyz, out_newxyz, idxbuf, radii2[si], ns);
        for (int b = 0; b < BATCH; b++) {
            gather_kernel<<<ncols, 160, 0, stream>>>(xyz, featT, out_newxyz, idxbuf, grouped, b, ns);
            const int l0 = si * 3, l1 = l0 + 1, l2 = l0 + 2;
            dim3 blk(128);  // 4 waves per block, one 64x32 macro-tile each
            gemm_kernel<0, 160><<<dim3(ncols / 128, Osz[l0] / 64), blk, 0, stream>>>(
                wbf[l0], grouped, actA, nullptr, nullptr, scb[l0],
                Osz[l0], ncols, ns, 0, b);
            gemm_kernel<0, 128><<<dim3(ncols / 128, Osz[l1] / 64), blk, 0, stream>>>(
                wbf[l1], actA, actB, nullptr, nullptr, scb[l1],
                Osz[l1], ncols, ns, 0, b);
            gemm_kernel<1, 128><<<dim3(ncols / 128, Osz[l2] / 64), blk, 0, stream>>>(
                wbf[l2], actB, nullptr, pooled, nullptr, scb[l2],
                Osz[l2], ncols, ns, si * 256, b);
        }
    }

    // ---- 5) pooled fp32 -> bf16, then final 512x512 GEMM + BN + ReLU ----
    pool2bf_kernel<<<(BATCH * SPTS * 512 + 255) / 256, 256, 0, stream>>>(
        pooled, pooledbf, BATCH * SPTS * 512);
    gemm_kernel<2, 512><<<dim3(BATCH * SPTS / 128, 512 / 64), dim3(128), 0, stream>>>(
        wbf[6], pooledbf, nullptr, nullptr, out_nf, scb[6],
        512, BATCH * SPTS, 1, 0, 0);
}
